// SimpleGCN_26431228739584
// MI455X (gfx1250) — matrix operand order repaired and checked by reading, not compiled, not run on hardware
//
#include <hip/hip_runtime.h>
#include <hip/hip_bf16.h>
#include <math.h>

typedef __attribute__((ext_vector_type(16))) __bf16 v16bf;
typedef __attribute__((ext_vector_type(8)))  float  v8f;

#define DH 128   // hidden dim (fixed by the model)

// ---------------------------------------------------------------------------
// A-fragment loader: 16x32 bf16 tile of row-major fp32 matrix A (ld = 128).
// ISA layout (16-bit A 16x32): lanes 0-15 -> M=0..15, K = {0..7,16..23};
// lanes 16-31 -> M=0..15, K = {8..15,24..31}; 2 K-values per VGPR.
// ---------------------------------------------------------------------------
__device__ __forceinline__ v16bf load_frag_a(const float* __restrict__ A,
                                             int row0, int k0, int lane, int M) {
    int m = row0 + (lane & 15);
    if (m >= M) m = M - 1;                      // clamp (EXEC must stay full)
    const float* p = A + (size_t)m * DH + k0 + ((lane >> 4) & 1) * 8;
    v16bf a;
#pragma unroll
    for (int i = 0; i < 8; ++i) {
        int kk = (i < 4) ? (2 * i) : (16 + 2 * (i - 4));
        float2 v = *reinterpret_cast<const float2*>(p + kk);
        a[2 * i]     = (__bf16)v.x;
        a[2 * i + 1] = (__bf16)v.y;
    }
    return a;
}

// ---------------------------------------------------------------------------
// B-fragment loader: 32x16 bf16 tile of row-major fp32 W (ld = 128).
// ISA layout: lanes 0-15 hold N=0..15 for K={0..7,16..23}, lanes 16-31 for K={8..15,24..31}.
// ---------------------------------------------------------------------------
__device__ __forceinline__ v16bf load_frag_b(const float* __restrict__ W,
                                             int k0, int n0, int lane) {
    int n = n0 + (lane & 15);
    int kb = k0 + ((lane >> 4) & 1) * 8;
    v16bf b;
#pragma unroll
    for (int i = 0; i < 8; ++i) {
        int kk = (i < 4) ? (2 * i) : (16 + 2 * (i - 4));
        b[2 * i]     = (__bf16)W[(size_t)(kb + kk)     * DH + n];
        b[2 * i + 1] = (__bf16)W[(size_t)(kb + kk + 1) * DH + n];
    }
    return b;
}

// ---------------------------------------------------------------------------
// out[M x 128] = A[M x 128] @ W[128 x 128] (+ bias).  One wave per 16x128 C
// tile; A fragments (4 K-steps) cached in VGPRs across all 8 N-tiles.
// 32x v_wmma_f32_16x16x32_bf16 per wave.
// ---------------------------------------------------------------------------
__global__ void __launch_bounds__(256)
gemm_wmma_128(const float* __restrict__ A, const float* __restrict__ W,
              const float* __restrict__ bias, float* __restrict__ out, int M) {
    int lane  = threadIdx.x & 31;
    int wave  = threadIdx.x >> 5;
    int mtile = blockIdx.x * (blockDim.x >> 5) + wave;
    if (mtile * 16 >= M) return;                // wave-uniform guard
    int row0 = mtile * 16;

    v16bf afrag[4];
#pragma unroll
    for (int k = 0; k < 4; ++k)
        afrag[k] = load_frag_a(A, row0, k * 32, lane, M);

#pragma unroll
    for (int nt = 0; nt < 8; ++nt) {
        v8f c = {};
#pragma unroll
        for (int k = 0; k < 4; ++k) {
            v16bf bfrag = load_frag_b(W, k * 32, nt * 16, lane);
            c = __builtin_amdgcn_wmma_f32_16x16x32_bf16(
                    false, afrag[k], false, bfrag, (short)0, c, false, false);
        }
        // C/D layout: VGPR r -> (M = r + 8*(lane>=16), N = lane&15)
        int n     = nt * 16 + (lane & 15);
        int mbase = row0 + ((lane >> 4) & 1) * 8;
        float bb  = bias ? bias[n] : 0.0f;
#pragma unroll
        for (int r = 0; r < 8; ++r) {
            int m = mbase + r;
            if (m < M) out[(size_t)m * DH + n] = c[r] + bb;
        }
    }
}

// --------------------------- normalization prep ----------------------------
__global__ void deg_init_kernel(float* deg, int N) {
    int i = blockIdx.x * blockDim.x + threadIdx.x;
    if (i < N) deg[i] = 1.0f;                   // self-loop weight
}

__global__ void deg_accum_kernel(const long long* __restrict__ col,
                                 const float* __restrict__ ew,
                                 float* deg, long long E) {
    long long e = (long long)blockIdx.x * blockDim.x + threadIdx.x;
    if (e < E) atomicAdd(&deg[(int)col[e]], ew[e]);
}

__global__ void dinv_kernel(const float* __restrict__ deg, float* dinv, int N) {
    int i = blockIdx.x * blockDim.x + threadIdx.x;
    if (i < N) { float d = deg[i]; dinv[i] = d > 0.0f ? rsqrtf(d) : 0.0f; }
}

__global__ void norm_kernel(const long long* __restrict__ row,
                            const long long* __restrict__ col,
                            const float* __restrict__ ew,
                            const float* __restrict__ dinv,
                            float* __restrict__ norm, long long E) {
    long long e = (long long)blockIdx.x * blockDim.x + threadIdx.x;
    if (e < E) norm[e] = dinv[(int)row[e]] * ew[e] * dinv[(int)col[e]];
}

// ------------------- aggregation: self-loop + bias init --------------------
// out[i,:] = bias + dinv[i]^2 * t[i,:]   (one wave per node, float4 per lane)
__global__ void agg_init_kernel(const float* __restrict__ t,
                                const float* __restrict__ dinv,
                                const float* __restrict__ bias,
                                float* __restrict__ out, int N) {
    int tid  = blockIdx.x * blockDim.x + threadIdx.x;
    int node = tid >> 5, lane = tid & 31;
    if (node >= N) return;
    float di = dinv[node];
    float s  = di * di;
    float4 v = *reinterpret_cast<const float4*>(t + (size_t)node * DH + lane * 4);
    const float4 b = *reinterpret_cast<const float4*>(bias + lane * 4);
    float4 o;
    o.x = b.x + s * v.x;  o.y = b.y + s * v.y;
    o.z = b.z + s * v.z;  o.w = b.w + s * v.w;
    *reinterpret_cast<float4*>(out + (size_t)node * DH + lane * 4) = o;
}

// --------------- aggregation: edge scatter (L2-resident atomics) -----------
// one wave per edge: gather b128, scale, 4x global_atomic_add_f32 per lane
__global__ void edge_scatter_kernel(const float* __restrict__ t,
                                    const long long* __restrict__ row,
                                    const long long* __restrict__ col,
                                    const float* __restrict__ norm,
                                    float* __restrict__ out, long long E) {
    long long tid = (long long)blockIdx.x * blockDim.x + threadIdx.x;
    long long e   = tid >> 5;
    int lane      = (int)(tid & 31);
    if (e >= E) return;
    int r = (int)row[e], c = (int)col[e];
    float w  = norm[e];
    float4 v = *reinterpret_cast<const float4*>(t + (size_t)r * DH + lane * 4);
    float* dst = out + (size_t)c * DH + lane * 4;
    atomicAdd(dst + 0, w * v.x);
    atomicAdd(dst + 1, w * v.y);
    atomicAdd(dst + 2, w * v.z);
    atomicAdd(dst + 3, w * v.w);
}

// ----------------- final head: sigmoid(h @ W2 + b2), d_out=1 ---------------
__global__ void head_kernel(const float* __restrict__ h,
                            const float* __restrict__ W2,
                            const float* __restrict__ b2,
                            float* __restrict__ out, int N) {
    int tid  = blockIdx.x * blockDim.x + threadIdx.x;
    int node = tid >> 5, lane = tid & 31;
    if (node >= N) return;
    float4 v = *reinterpret_cast<const float4*>(h + (size_t)node * DH + lane * 4);
    float4 w = *reinterpret_cast<const float4*>(W2 + lane * 4);
    float s  = v.x * w.x + v.y * w.y + v.z * w.z + v.w * w.w;
#pragma unroll
    for (int off = 16; off > 0; off >>= 1)
        s += __shfl_xor(s, off, 32);            // wave32 reduction
    if (lane == 0) out[node] = 1.0f / (1.0f + __expf(-(s + b2[0])));
}

// ---------------------------------------------------------------------------
extern "C" void kernel_launch(void* const* d_in, const int* in_sizes, int n_in,
                              void* d_out, int out_size, void* d_ws, size_t ws_size,
                              hipStream_t stream) {
    const float*     x  = (const float*)d_in[0];
    const long long* ei = (const long long*)d_in[1];   // int64 (2, E)
    const float*     ea = (const float*)d_in[2];
    const float*     W1 = (const float*)d_in[3];
    const float*     b1 = (const float*)d_in[4];
    const float*     Wg = (const float*)d_in[5];       // (3,128,128)
    const float*     bg = (const float*)d_in[6];       // (3,128)
    const float*     W2 = (const float*)d_in[7];       // (128,1)
    const float*     b2 = (const float*)d_in[8];

    const int       N = in_sizes[0] / DH;
    const long long E = in_sizes[2];
    const int       L = in_sizes[5] / (DH * DH);

    const long long* row = ei;
    const long long* col = ei + E;

    // workspace carve-up
    float* ws   = (float*)d_ws;
    size_t off  = 0;
    float* deg  = ws + off;  off += (size_t)N;
    float* dinv = ws + off;  off += (size_t)N;
    float* nrm  = ws + off;  off += (size_t)E;
    off = (off + 3) & ~(size_t)3;
    float* hA   = ws + off;  off += (size_t)N * DH;
    float* hB   = ws + off;

    const int T = 256;
    // 1) normalization coefficients
    deg_init_kernel<<<(N + T - 1) / T, T, 0, stream>>>(deg, N);
    deg_accum_kernel<<<(int)((E + T - 1) / T), T, 0, stream>>>(col, ea, deg, E);
    dinv_kernel<<<(N + T - 1) / T, T, 0, stream>>>(deg, dinv, N);
    norm_kernel<<<(int)((E + T - 1) / T), T, 0, stream>>>(row, col, ea, dinv, nrm, E);

    // 2) input projection: hA = x @ W1 + b1
    int mtiles = (N + 15) / 16;
    int gblk   = (mtiles + 7) / 8;                     // 8 waves/block
    gemm_wmma_128<<<gblk, T, 0, stream>>>(x, W1, b1, hA, N);

    // 3) GCN layers
    long long escat = (E * 32 + T - 1) / T;
    int       nwarp = (N * 32 + T - 1) / T;
    for (int l = 0; l < L; ++l) {
        gemm_wmma_128<<<gblk, T, 0, stream>>>(hA, Wg + (size_t)l * DH * DH,
                                              nullptr, hB, N);
        agg_init_kernel<<<nwarp, T, 0, stream>>>(hB, dinv, bg + (size_t)l * DH, hA, N);
        edge_scatter_kernel<<<(int)escat, T, 0, stream>>>(hB, row, col, nrm, hA, E);
    }

    // 4) head: sigmoid(hA @ W2 + b2)
    head_kernel<<<nwarp, T, 0, stream>>>(hA, W2, b2, (float*)d_out, N);
}
